// SwinTransformerBlock_83013127897162
// MI455X (gfx1250) — matrix operand order repaired and checked by reading, not compiled, not run on hardware
//
#include <hip/hip_runtime.h>
#include <hip/hip_bf16.h>

typedef _Float16 half_t;
typedef __attribute__((ext_vector_type(16))) _Float16 v16h;
typedef __attribute__((ext_vector_type(8)))  _Float16 v8h;
typedef __attribute__((ext_vector_type(8)))  float    v8f;

#define DEVINL __device__ __forceinline__

// ---------------- WMMA fragment helpers (gfx1250, wave32) ----------------
// A: 16x32 f16, row-major [M][ldk]. lane: m=l%16; halves 0..7 -> k0..k0+7,
// halves 8..15 -> k0+16..k0+23, k0 = (l>=16)*8.
DEVINL v16h frag_a(const half_t* tile, int ldk, int lane) {
  int m = lane & 15;
  int g = lane >> 4;
  const half_t* p = tile + m * ldk + g * 8;
  v8h lo = *(const v8h*)(p);
  v8h hi = *(const v8h*)(p + 16);
  v16h r;
#pragma unroll
  for (int i = 0; i < 8; ++i) { r[i] = lo[i]; r[i + 8] = hi[i]; }
  return r;
}

// B: 32x16 f16 stored transposed [N][ldk]. lane: n=l%16; halves 0..7 -> k0..k0+7,
// halves 8..15 -> k0+16..k0+23, k0 = (l>=16)*8.
DEVINL v16h frag_b(const half_t* tileT, int ldk, int lane) {
  int n = lane & 15;
  int g = lane >> 4;
  const half_t* p = tileT + n * ldk + g * 8;
  v8h lo = *(const v8h*)(p);
  v8h hi = *(const v8h*)(p + 16);
  v16h r;
#pragma unroll
  for (int i = 0; i < 8; ++i) { r[i] = lo[i]; r[i + 8] = hi[i]; }
  return r;
}

DEVINL v8f gemm_tile(const half_t* A, int lda, const half_t* B, int ldb,
                     int ksteps, int lane) {
  v8f acc = {0.f, 0.f, 0.f, 0.f, 0.f, 0.f, 0.f, 0.f};
  for (int ks = 0; ks < ksteps; ++ks) {
    v16h a = frag_a(A + ks * 32, lda, lane);
    v16h b = frag_b(B + ks * 32, ldb, lane);
    acc = __builtin_amdgcn_wmma_f32_16x16x32_f16(false, a, false, b, (short)0,
                                                 acc, false, false);
  }
  return acc;
}

// ---------------- constants ----------------
constexpr int Bn = 32, Hd = 64, Wd = 64, Cc = 192;
constexpr int WS = 8, SHIFT = 4, NH = 6, HD = 32, Nt = 64, HID = 384;
constexpr int NWIN = Bn * (Hd / WS) * (Wd / WS);         // 2048 windows
constexpr float SCALE = 0.17677669529663687f;            // 32^-0.5

// workspace layout (bytes)
constexpr size_t OFF_H   = 0;                            // f32 [B,H,W,C]   100663296
constexpr size_t OFF_Q   = 100663296;                    // f16 Q           50331648
constexpr size_t OFF_K   = OFF_Q + 50331648;             // f16 K
constexpr size_t OFF_VT  = OFF_K + 50331648;             // f16 V^T
constexpr size_t OFF_XW  = OFF_VT + 50331648;            // f16 Xwin / attnout / LN2-out
constexpr size_t OFF_HID = OFF_Q;                        // f16 hid (reuses Q+K)
constexpr size_t OFF_W   = OFF_XW + 50331648;            // f16 transposed weights

// ---------------- weight transpose + f32->f16 ----------------
__global__ void k_transpose(const float* __restrict__ src, half_t* __restrict__ dst,
                            int K, int N) {
  int idx = blockIdx.x * 256 + threadIdx.x;
  if (idx < K * N) {
    int n = idx / K, k = idx - n * K;
    dst[idx] = (half_t)src[k * N + n];   // dst[n][k]
  }
}

// ---------------- LN1 + cyclic shift + window partition -> f16 ----------------
__global__ void k_ln1(const float* __restrict__ x, const float* __restrict__ g,
                      const float* __restrict__ be, half_t* __restrict__ xw) {
  int lane = threadIdx.x & 31;
  int wv = threadIdx.x >> 5;
  int t = blockIdx.x * 8 + wv;                 // dest token [0, 131072)
  int wi = t >> 6, tok = t & 63;
  int b = wi >> 6, widx = wi & 63;
  int wh = widx >> 3, ww = widx & 7;
  int r = tok >> 3, c = tok & 7;
  int sh = ((wh << 3) + r + SHIFT) & 63;
  int sw = ((ww << 3) + c + SHIFT) & 63;
  const float* src = x + (((size_t)(b * 64 + sh)) * 64 + sw) * Cc;
  float v[6]; float s = 0.f;
#pragma unroll
  for (int i = 0; i < 6; ++i) { v[i] = src[i * 32 + lane]; s += v[i]; }
#pragma unroll
  for (int m = 16; m >= 1; m >>= 1) s += __shfl_xor(s, m);
  float mean = s * (1.0f / 192.0f);
  float q = 0.f;
#pragma unroll
  for (int i = 0; i < 6; ++i) { float d = v[i] - mean; q += d * d; }
#pragma unroll
  for (int m = 16; m >= 1; m >>= 1) q += __shfl_xor(q, m);
  float rstd = rsqrtf(q * (1.0f / 192.0f) + 1e-5f);
  half_t* dst = xw + (size_t)t * Cc;
#pragma unroll
  for (int i = 0; i < 6; ++i) {
    int ch = i * 32 + lane;
    dst[ch] = (half_t)((v[i] - mean) * rstd * g[ch] + be[ch]);
  }
}

// ---------------- QKV GEMM: [64,192] @ [192,576] per window ----------------
__global__ void k_qkv(const half_t* __restrict__ xw, const half_t* __restrict__ wT,
                      const float* __restrict__ bias, half_t* __restrict__ Q,
                      half_t* __restrict__ Kb, half_t* __restrict__ Vt) {
  int lane = threadIdx.x & 31;
  int wv = threadIdx.x >> 5;
  int wi = blockIdx.x;
  int g = lane >> 4, cn = lane & 15;
  const half_t* A = xw + (size_t)wi * Nt * Cc;
  for (int t = wv; t < 144; t += 8) {          // mt invariant -> A frags hoisted
    int mt = t & 3, nt = t >> 2;
    if (t + 8 < 144)                           // prefetch next B tile (this wave)
      __builtin_prefetch(wT + ((t + 8) >> 2) * 16 * Cc, 0, 2);
    v8f acc = gemm_tile(A + mt * 16 * Cc, Cc, wT + nt * 16 * Cc, Cc, 6, lane);
    // tile-uniform decomposition of the output column block (SALU, no exec churn)
    int sec = nt / 12;                         // 0=q 1=k 2=v
    int ntl = nt - sec * 12;
    int h = ntl >> 1;
    int d = ((ntl & 1) << 4) + cn;
    float bn = bias[nt * 16 + cn];
    size_t base = ((size_t)(wi * NH + h)) * (Nt * HD);
    if (sec == 0) {
#pragma unroll
      for (int j = 0; j < 8; ++j) {
        int m = mt * 16 + g * 8 + j;
        Q[base + m * HD + d] = (half_t)((acc[j] + bn) * SCALE);
      }
    } else if (sec == 1) {
#pragma unroll
      for (int j = 0; j < 8; ++j) {
        int m = mt * 16 + g * 8 + j;
        Kb[base + m * HD + d] = (half_t)(acc[j] + bn);
      }
    } else {
#pragma unroll
      for (int j = 0; j < 8; ++j) {
        int m = mt * 16 + g * 8 + j;
        Vt[base + d * Nt + m] = (half_t)(acc[j] + bn);   // transposed
      }
    }
  }
}

DEVINL int regf(int t) { return t < 56 ? 0 : (t < 60 ? 1 : 2); }

// ---------------- attention per (window, head) ----------------
__global__ void k_attn(const half_t* __restrict__ Q, const half_t* __restrict__ Kb,
                       const half_t* __restrict__ Vt, const float* __restrict__ relb,
                       half_t* __restrict__ outp) {
  __shared__ half_t P[Nt * Nt];                // probs, f16, A-fragment layout
  __shared__ float relS[225];                  // this head's rel-bias column
  int lane = threadIdx.x & 31;
  int w = threadIdx.x >> 5;                    // 4 waves, one m-tile each
  int bh = blockIdx.x;
  int wi = bh / NH, h = bh - wi * NH;
  size_t base = (size_t)bh * (Nt * HD);
  int widx = wi & 63, wh = widx >> 3, ww = widx & 7;
  int g = lane >> 4, cn = lane & 15;

  for (int i = threadIdx.x; i < 225; i += 128) relS[i] = relb[i * NH + h];

  // scores = (q*scale) @ k^T   (K-dim = 32, 1 WMMA per 16x16 tile)
  v16h aq = frag_a(Q + base + w * 16 * HD, HD, lane);
  v8f acc[4];
#pragma unroll
  for (int nt = 0; nt < 4; ++nt) {
    v8f z = {0.f, 0.f, 0.f, 0.f, 0.f, 0.f, 0.f, 0.f};
    v16h bk = frag_b(Kb + base + nt * 16 * HD, HD, lane);
    acc[nt] = __builtin_amdgcn_wmma_f32_16x16x32_f16(false, aq, false, bk,
                                                     (short)0, z, false, false);
  }
  __syncthreads();                             // relS ready

  // + rel-pos bias + shift mask, row softmax, write probs to LDS
#pragma unroll
  for (int j = 0; j < 8; ++j) {
    int i = w * 16 + g * 8 + j;                // row token
    int ri = i >> 3, ci = i & 7;
    int regi = regf(wh * 8 + ri) * 3 + regf(ww * 8 + ci);
    float vals[4]; float rowmax = -1e30f;
#pragma unroll
    for (int nt = 0; nt < 4; ++nt) {
      int jj = nt * 16 + cn;                   // col token
      int rj = jj >> 3, cj = jj & 7;
      int ridx = (ri - rj + 7) * 15 + (ci - cj + 7);
      float bia = relS[ridx];
      int regj = regf(wh * 8 + rj) * 3 + regf(ww * 8 + cj);
      float msk = (regi != regj) ? -100.0f : 0.0f;
      vals[nt] = acc[nt][j] + bia + msk;
      rowmax = fmaxf(rowmax, vals[nt]);
    }
#pragma unroll
    for (int s = 8; s >= 1; s >>= 1) rowmax = fmaxf(rowmax, __shfl_xor(rowmax, s));
    float sum = 0.f;
#pragma unroll
    for (int nt = 0; nt < 4; ++nt) { vals[nt] = __expf(vals[nt] - rowmax); sum += vals[nt]; }
#pragma unroll
    for (int s = 8; s >= 1; s >>= 1) sum += __shfl_xor(sum, s);
    float inv = 1.0f / sum;
#pragma unroll
    for (int nt = 0; nt < 4; ++nt)
      P[i * Nt + nt * 16 + cn] = (half_t)(vals[nt] * inv);
  }
  __syncthreads();

  // out = P @ V   (K-dim = 64 -> 2 WMMA per tile; B from transposed V)
#pragma unroll
  for (int nt2 = 0; nt2 < 2; ++nt2) {
    v8f o = {0.f, 0.f, 0.f, 0.f, 0.f, 0.f, 0.f, 0.f};
#pragma unroll
    for (int ks = 0; ks < 2; ++ks) {
      v16h a = frag_a(&P[w * 16 * Nt + ks * 32], Nt, lane);
      v16h b = frag_b(Vt + base + nt2 * 16 * Nt + ks * 32, Nt, lane);
      o = __builtin_amdgcn_wmma_f32_16x16x32_f16(false, a, false, b, (short)0,
                                                 o, false, false);
    }
#pragma unroll
    for (int j = 0; j < 8; ++j) {
      int m = w * 16 + g * 8 + j;
      int d = nt2 * 16 + cn;
      outp[(size_t)wi * Nt * Cc + m * Cc + h * HD + d] = (half_t)o[j];
    }
  }
}

// ---------------- proj GEMM + window-reverse + un-shift + residual ----------------
__global__ void k_proj(const half_t* __restrict__ ao, const half_t* __restrict__ wT,
                       const float* __restrict__ pb, const float* __restrict__ x,
                       float* __restrict__ hb) {
  int lane = threadIdx.x & 31;
  int wv = threadIdx.x >> 5;
  int wi = blockIdx.x;
  const half_t* A = ao + (size_t)wi * Nt * Cc;
  int b = wi >> 6, widx = wi & 63, wh = widx >> 3, ww = widx & 7;
  for (int t = wv; t < 48; t += 8) {           // 4 m-tiles x 12 n-tiles
    int mt = t & 3, nt = t >> 2;
    if (t + 8 < 48)
      __builtin_prefetch(wT + ((t + 8) >> 2) * 16 * Cc, 0, 2);
    v8f acc = gemm_tile(A + mt * 16 * Cc, Cc, wT + nt * 16 * Cc, Cc, 6, lane);
    int n = nt * 16 + (lane & 15);
    int g = lane >> 4;
    float bn = pb[n];
#pragma unroll
    for (int j = 0; j < 8; ++j) {
      int tok = mt * 16 + g * 8 + j;
      int r = tok >> 3, c = tok & 7;
      int oh = ((wh << 3) + r + SHIFT) & 63;
      int ow = ((ww << 3) + c + SHIFT) & 63;
      size_t idx = (((size_t)(b * 64 + oh)) * 64 + ow) * Cc + n;
      hb[idx] = x[idx] + acc[j] + bn;
    }
  }
}

// ---------------- LN2 ----------------
__global__ void k_ln2(const float* __restrict__ hsrc, const float* __restrict__ g,
                      const float* __restrict__ be, half_t* __restrict__ mout) {
  int lane = threadIdx.x & 31;
  int wv = threadIdx.x >> 5;
  int t = blockIdx.x * 8 + wv;
  const float* src = hsrc + (size_t)t * Cc;
  float v[6]; float s = 0.f;
#pragma unroll
  for (int i = 0; i < 6; ++i) { v[i] = src[i * 32 + lane]; s += v[i]; }
#pragma unroll
  for (int m = 16; m >= 1; m >>= 1) s += __shfl_xor(s, m);
  float mean = s * (1.0f / 192.0f);
  float q = 0.f;
#pragma unroll
  for (int i = 0; i < 6; ++i) { float d = v[i] - mean; q += d * d; }
#pragma unroll
  for (int m = 16; m >= 1; m >>= 1) q += __shfl_xor(q, m);
  float rstd = rsqrtf(q * (1.0f / 192.0f) + 1e-5f);
  half_t* dst = mout + (size_t)t * Cc;
#pragma unroll
  for (int i = 0; i < 6; ++i) {
    int ch = i * 32 + lane;
    dst[ch] = (half_t)((v[i] - mean) * rstd * g[ch] + be[ch]);
  }
}

// ---------------- MLP1 GEMM + exact GELU ----------------
__global__ void k_mlp1(const half_t* __restrict__ A, const half_t* __restrict__ wT,
                       const float* __restrict__ b1, half_t* __restrict__ hid) {
  int lane = threadIdx.x & 31;
  int wv = threadIdx.x >> 5;
  size_t rb = (size_t)blockIdx.x * 128;
  for (int t = wv; t < 192; t += 8) {          // 8 m-tiles x 24 n-tiles
    int mt = t & 7, nt = t >> 3;
    if (t + 8 < 192)
      __builtin_prefetch(wT + ((t + 8) >> 3) * 16 * Cc, 0, 2);
    v8f acc = gemm_tile(A + (rb + mt * 16) * Cc, Cc, wT + nt * 16 * Cc, Cc, 6, lane);
    int n = nt * 16 + (lane & 15);
    int g = lane >> 4;
    float bn = b1[n];
#pragma unroll
    for (int j = 0; j < 8; ++j) {
      size_t row = rb + mt * 16 + g * 8 + j;
      float v = acc[j] + bn;
      v = 0.5f * v * (1.0f + erff(v * 0.70710678118f));
      hid[row * HID + n] = (half_t)v;
    }
  }
}

// ---------------- MLP2 GEMM + residual -> f32 out ----------------
__global__ void k_mlp2(const half_t* __restrict__ A, const half_t* __restrict__ wT,
                       const float* __restrict__ b2, const float* __restrict__ hb,
                       float* __restrict__ out) {
  int lane = threadIdx.x & 31;
  int wv = threadIdx.x >> 5;
  size_t rb = (size_t)blockIdx.x * 128;
  for (int t = wv; t < 96; t += 8) {           // 8 m-tiles x 12 n-tiles
    int mt = t & 7, nt = t >> 3;
    if (t + 8 < 96)
      __builtin_prefetch(wT + ((t + 8) >> 3) * 16 * HID, 0, 2);
    v8f acc = gemm_tile(A + (rb + mt * 16) * HID, HID, wT + nt * 16 * HID, HID, 12, lane);
    int n = nt * 16 + (lane & 15);
    int g = lane >> 4;
    float bn = b2[n];
#pragma unroll
    for (int j = 0; j < 8; ++j) {
      size_t row = rb + mt * 16 + g * 8 + j;
      out[row * Cc + n] = hb[row * Cc + n] + acc[j] + bn;
    }
  }
}

extern "C" void kernel_launch(void* const* d_in, const int* in_sizes, int n_in,
                              void* d_out, int out_size, void* d_ws, size_t ws_size,
                              hipStream_t stream) {
  (void)in_sizes; (void)n_in; (void)out_size; (void)ws_size;
  const float* x      = (const float*)d_in[0];
  const float* n1g    = (const float*)d_in[1];
  const float* n1b    = (const float*)d_in[2];
  const float* qkv_w  = (const float*)d_in[3];
  const float* qkv_b  = (const float*)d_in[4];
  const float* proj_w = (const float*)d_in[5];
  const float* proj_b = (const float*)d_in[6];
  const float* relb   = (const float*)d_in[7];
  const float* n2g    = (const float*)d_in[8];
  const float* n2b    = (const float*)d_in[9];
  const float* w1     = (const float*)d_in[10];
  const float* b1     = (const float*)d_in[11];
  const float* w2     = (const float*)d_in[12];
  const float* b2     = (const float*)d_in[13];

  char* ws = (char*)d_ws;
  float*  hb    = (float*)(ws + OFF_H);
  half_t* Q     = (half_t*)(ws + OFF_Q);
  half_t* Kb    = (half_t*)(ws + OFF_K);
  half_t* Vt    = (half_t*)(ws + OFF_VT);
  half_t* Xw    = (half_t*)(ws + OFF_XW);   // Xwin -> attnout -> LN2-out
  half_t* hid   = (half_t*)(ws + OFF_HID);
  half_t* qkvT  = (half_t*)(ws + OFF_W);
  half_t* projT = qkvT + 576 * 192;
  half_t* w1T   = projT + 192 * 192;
  half_t* w2T   = w1T + 384 * 192;

  k_transpose<<<(576 * 192 + 255) / 256, 256, 0, stream>>>(qkv_w, qkvT, 192, 576);
  k_transpose<<<(192 * 192 + 255) / 256, 256, 0, stream>>>(proj_w, projT, 192, 192);
  k_transpose<<<(384 * 192 + 255) / 256, 256, 0, stream>>>(w1, w1T, 192, 384);
  k_transpose<<<(384 * 192 + 255) / 256, 256, 0, stream>>>(w2, w2T, 384, 192);

  k_ln1 <<<16384, 256, 0, stream>>>(x, n1g, n1b, Xw);
  k_qkv <<<NWIN, 256, 0, stream>>>(Xw, qkvT, qkv_b, Q, Kb, Vt);
  k_attn<<<NWIN * NH, 128, 0, stream>>>(Q, Kb, Vt, relb, Xw);
  k_proj<<<NWIN, 256, 0, stream>>>(Xw, projT, proj_b, x, hb);
  k_ln2 <<<16384, 256, 0, stream>>>(hb, n2g, n2b, Xw);
  k_mlp1<<<1024, 256, 0, stream>>>(Xw, w1T, b1, hid);
  k_mlp2<<<1024, 256, 0, stream>>>(hid, w2T, b2, hb, (float*)d_out);
}
